// SelfAttention_37529424232916
// MI455X (gfx1250) — compile-verified, matrix operand order checked by reading, not run on hardware
//
#include <hip/hip_runtime.h>
#include <hip/hip_bf16.h>

typedef __attribute__((ext_vector_type(16))) __bf16 v16bf;
typedef __attribute__((ext_vector_type(8)))  float  v8f;

#define B_LEN 4
#define S_LEN 2048
#define D_LEN 1024
#define M_TOT (B_LEN * S_LEN)   // 8192 rows
#define NCHUNK 32
#define CHUNK  (S_LEN / NCHUNK) // 64

// ---------------------------------------------------------------------------
// Kernel 1: fused QKV projection, Y = x @ W^T via bf16 hi/lo split WMMA.
// x fp32 is split as x = hi + lo (both bf16); product uses 3 WMMAs:
//   hi*hi + hi*lo + lo*hi (lo*lo ~ 2^-32 relative, dropped) -> ~fp32 accuracy
// with v_wmma_f32_16x16x32_bf16 (fp32 accumulate).
// ---------------------------------------------------------------------------
__global__ __launch_bounds__(256)
void qkv_gemm_kernel(const float* __restrict__ x,
                     const float* __restrict__ Wq,
                     const float* __restrict__ Wk,
                     const float* __restrict__ Wv,
                     float* __restrict__ q,
                     float* __restrict__ kk,
                     float* __restrict__ vv)
{
    const float* W = (blockIdx.z == 0) ? Wq : (blockIdx.z == 1) ? Wk : Wv;
    float*       Y = (blockIdx.z == 0) ? q  : (blockIdx.z == 1) ? kk : vv;

    __shared__ float lA[32][36];    // 32 rows of x  x 32 K   (padded)
    __shared__ float lB[128][36];   // 128 rows of W x 32 K   (padded)

    const int tid  = threadIdx.x;
    const int wave = tid >> 5;
    const int lane = tid & 31;
    const int wm   = wave >> 2;          // 0..1  (M sub-tile)
    const int wn   = wave & 3;           // 0..3  (N sub-tile)
    const int l16  = lane & 15;
    const int hi16 = (lane >> 4) & 1;    // lane >= 16

    const int m_base = blockIdx.x * 32;
    const int n_base = blockIdx.y * 128;

    v8f acc0 = {};
    v8f acc1 = {};

    for (int k0 = 0; k0 < D_LEN; k0 += 32) {
        // stage A tile: 32x32 f32, one float4 per thread
        {
            const int row = tid >> 3;
            const int kc  = (tid & 7) * 4;
            const float4 a = *(const float4*)(x + (size_t)(m_base + row) * D_LEN + k0 + kc);
            lA[row][kc + 0] = a.x; lA[row][kc + 1] = a.y;
            lA[row][kc + 2] = a.z; lA[row][kc + 3] = a.w;
        }
        // stage B tile: 128x32 f32 (W[n][k]), four float4 per thread
        #pragma unroll
        for (int i = 0; i < 4; ++i) {
            const int row = (tid >> 3) + 32 * i;
            const int kc  = (tid & 7) * 4;
            const float4 b = *(const float4*)(W + (size_t)(n_base + row) * D_LEN + k0 + kc);
            lB[row][kc + 0] = b.x; lB[row][kc + 1] = b.y;
            lB[row][kc + 2] = b.z; lB[row][kc + 3] = b.w;
        }
        __syncthreads();

        // build bf16 hi/lo fragments per ISA 16-bit A layout:
        // element e -> k = e + (e&8) + 8*(lane>=16); row/col = lane&15
        v16bf ah, al, bh0, bl0, bh1, bl1;
        const int arow = wm * 16 + l16;
        const int nc0  = wn * 32 + l16;
        const int nc1  = nc0 + 16;
        #pragma unroll
        for (int e = 0; e < 16; ++e) {
            const int kc = e + (e & 8) + (hi16 << 3);
            float  av = lA[arow][kc];
            __bf16 h  = (__bf16)av;
            __bf16 l  = (__bf16)(av - (float)h);
            ah[e] = h; al[e] = l;

            float b0 = lB[nc0][kc];
            h = (__bf16)b0; l = (__bf16)(b0 - (float)h);
            bh0[e] = h; bl0[e] = l;

            float b1 = lB[nc1][kc];
            h = (__bf16)b1; l = (__bf16)(b1 - (float)h);
            bh1[e] = h; bl1[e] = l;
        }

        acc0 = __builtin_amdgcn_wmma_f32_16x16x32_bf16(false, ah, false, bh0, (short)0, acc0, false, false);
        acc0 = __builtin_amdgcn_wmma_f32_16x16x32_bf16(false, ah, false, bl0, (short)0, acc0, false, false);
        acc0 = __builtin_amdgcn_wmma_f32_16x16x32_bf16(false, al, false, bh0, (short)0, acc0, false, false);

        acc1 = __builtin_amdgcn_wmma_f32_16x16x32_bf16(false, ah, false, bh1, (short)0, acc1, false, false);
        acc1 = __builtin_amdgcn_wmma_f32_16x16x32_bf16(false, ah, false, bl1, (short)0, acc1, false, false);
        acc1 = __builtin_amdgcn_wmma_f32_16x16x32_bf16(false, al, false, bh1, (short)0, acc1, false, false);

        __syncthreads();
    }

    // C layout: VGPR r, lanes 0-15 -> (M=r, N=lane); lanes 16-31 -> (M=r+8, N=lane-16)
    #pragma unroll
    for (int r = 0; r < 8; ++r) {
        const int m   = r + (hi16 << 3);
        const size_t row = (size_t)(m_base + wm * 16 + m) * D_LEN;
        Y[row + n_base + wn * 32 + l16]      = acc0[r];
        Y[row + n_base + wn * 32 + 16 + l16] = acc1[r];
    }
}

// ---------------------------------------------------------------------------
// Shared-memory radix-2 DIT FFT of length 1024 (block = 128 threads).
// Input must already be bit-reverse scattered into `data`.
// ---------------------------------------------------------------------------
__device__ __forceinline__ float2 cmul(float2 a, float2 b) {
    return make_float2(a.x * b.x - a.y * b.y, a.x * b.y + a.y * b.x);
}

__device__ void block_fft_1024(float2* data, const float2* tw, int tid)
{
    #pragma unroll
    for (int len = 2; len <= 1024; len <<= 1) {
        const int half = len >> 1;
        const int tstr = 1024 / len;
        for (int b = tid; b < 512; b += 128) {
            const int g  = b / half;
            const int j  = b & (half - 1);
            const int i0 = g * len + j;
            const int i1 = i0 + half;
            const float2 w = tw[j * tstr];
            const float2 u = data[i0];
            const float2 t = cmul(w, data[i1]);
            data[i0] = make_float2(u.x + t.x, u.y + t.y);
            data[i1] = make_float2(u.x - t.x, u.y - t.y);
        }
        __syncthreads();
    }
}

__device__ __forceinline__ void init_twiddles(float2* tw, int tid)
{
    const float w0 = -6.283185307179586f / 1024.0f;
    for (int t = tid; t < 512; t += 128) {
        float s, c;
        sincosf(w0 * (float)t, &s, &c);
        tw[t] = make_float2(c, s);
    }
}

// ---------------------------------------------------------------------------
// Kernel 2: per-row forward FFTs. fq = fft(q); p = fft(k) * fft(v).
// One block per (b,s) row.
// ---------------------------------------------------------------------------
__global__ __launch_bounds__(128)
void fft_fwd_kernel(const float* __restrict__ q,
                    const float* __restrict__ k,
                    const float* __restrict__ v,
                    float2* __restrict__ fq,
                    float2* __restrict__ p)
{
    __shared__ float2 data[1024];
    __shared__ float2 tw[512];
    const int    tid = threadIdx.x;
    const size_t row = (size_t)blockIdx.x * D_LEN;

    init_twiddles(tw, tid);
    __syncthreads();

    // --- fft(q) -> fq ---
    for (int i = tid; i < 1024; i += 128)
        data[__brev((unsigned)i) >> 22] = make_float2(q[row + i], 0.0f);
    __syncthreads();
    block_fft_1024(data, tw, tid);
    for (int i = tid; i < 1024; i += 128) fq[row + i] = data[i];
    __syncthreads();

    // --- fft(k) -> registers ---
    for (int i = tid; i < 1024; i += 128)
        data[__brev((unsigned)i) >> 22] = make_float2(k[row + i], 0.0f);
    __syncthreads();
    block_fft_1024(data, tw, tid);
    float2 fkr[8];
    #pragma unroll
    for (int t = 0; t < 8; ++t) fkr[t] = data[tid + 128 * t];
    __syncthreads();

    // --- fft(v); p = fk * fv ---
    for (int i = tid; i < 1024; i += 128)
        data[__brev((unsigned)i) >> 22] = make_float2(v[row + i], 0.0f);
    __syncthreads();
    block_fft_1024(data, tw, tid);
    #pragma unroll
    for (int t = 0; t < 8; ++t) {
        const int i = tid + 128 * t;
        p[row + i] = cmul(fkr[t], data[i]);
    }
}

// ---------------------------------------------------------------------------
// Kernels 3-5: blocked prefix scan of p over the sequence axis (per b,d).
// ---------------------------------------------------------------------------
__global__ __launch_bounds__(256)
void chunk_sum_kernel(const float2* __restrict__ p, float2* __restrict__ csum)
{
    const int gid = blockIdx.x * blockDim.x + threadIdx.x;  // (b, c, d)
    if (gid >= B_LEN * NCHUNK * D_LEN) return;
    const int d = gid & (D_LEN - 1);
    const int c = (gid >> 10) & (NCHUNK - 1);
    const int b = gid >> 15;
    float2 acc = make_float2(0.0f, 0.0f);
    const size_t base = ((size_t)b * S_LEN + (size_t)c * CHUNK) * D_LEN + d;
    for (int s = 0; s < CHUNK; ++s) {
        if (s + 8 < CHUNK) __builtin_prefetch(p + base + (size_t)(s + 8) * D_LEN, 0, 1);
        const float2 t = p[base + (size_t)s * D_LEN];
        acc.x += t.x; acc.y += t.y;
    }
    csum[gid] = acc;
}

__global__ __launch_bounds__(256)
void chunk_scan_kernel(float2* __restrict__ csum)
{
    const int gid = blockIdx.x * blockDim.x + threadIdx.x;  // (b, d)
    if (gid >= B_LEN * D_LEN) return;
    const int d = gid & (D_LEN - 1);
    const int b = gid >> 10;
    float2 run = make_float2(0.0f, 0.0f);
    for (int c = 0; c < NCHUNK; ++c) {
        const size_t idx = ((size_t)b * NCHUNK + c) * D_LEN + d;
        const float2 t = csum[idx];
        csum[idx] = run;             // exclusive prefix, in place
        run.x += t.x; run.y += t.y;
    }
}

__global__ __launch_bounds__(256)
void scan_apply_kernel(float2* __restrict__ p, const float2* __restrict__ csum)
{
    const int gid = blockIdx.x * blockDim.x + threadIdx.x;  // (b, c, d)
    if (gid >= B_LEN * NCHUNK * D_LEN) return;
    const int d = gid & (D_LEN - 1);
    const int c = (gid >> 10) & (NCHUNK - 1);
    const int b = gid >> 15;
    float2 run = csum[gid];
    const size_t base = ((size_t)b * S_LEN + (size_t)c * CHUNK) * D_LEN + d;
    for (int s = 0; s < CHUNK; ++s) {
        const size_t idx = base + (size_t)s * D_LEN;
        const float2 t = p[idx];
        run.x += t.x; run.y += t.y;
        p[idx] = run;                // p becomes mem (inclusive scan), in place
    }
}

// ---------------------------------------------------------------------------
// Kernel 6: out = Re(ifft(mem * conj(fq))) = Re(fft(conj(mem) * fq)) / N
// ---------------------------------------------------------------------------
__global__ __launch_bounds__(128)
void ifft_kernel(const float2* __restrict__ mem,
                 const float2* __restrict__ fq,
                 float* __restrict__ out)
{
    __shared__ float2 data[1024];
    __shared__ float2 tw[512];
    const int    tid = threadIdx.x;
    const size_t row = (size_t)blockIdx.x * D_LEN;

    init_twiddles(tw, tid);
    __syncthreads();

    for (int i = tid; i < 1024; i += 128) {
        const float2 m = mem[row + i];
        const float2 f = fq[row + i];
        // conj(m) * f
        const float2 z = make_float2(m.x * f.x + m.y * f.y, m.x * f.y - m.y * f.x);
        data[__brev((unsigned)i) >> 22] = z;
    }
    __syncthreads();
    block_fft_1024(data, tw, tid);
    const float inv = 1.0f / 1024.0f;
    for (int i = tid; i < 1024; i += 128) out[row + i] = data[i].x * inv;
}

// ---------------------------------------------------------------------------
// Host launcher
// ---------------------------------------------------------------------------
extern "C" void kernel_launch(void* const* d_in, const int* in_sizes, int n_in,
                              void* d_out, int out_size, void* d_ws, size_t ws_size,
                              hipStream_t stream) {
    (void)in_sizes; (void)n_in; (void)out_size; (void)ws_size;
    const float* x  = (const float*)d_in[0];
    const float* Wq = (const float*)d_in[1];
    const float* Wk = (const float*)d_in[2];
    const float* Wv = (const float*)d_in[3];
    float*       out = (float*)d_out;

    // workspace layout
    const size_t MR = (size_t)M_TOT * D_LEN;   // 8M elements per fp32 plane
    float*  q    = (float*)d_ws;               // 32 MB
    float*  kk   = q + MR;                     // 32 MB
    float*  vv   = kk + MR;                    // 32 MB
    float2* fq   = (float2*)(vv + MR);         // 64 MB
    float2* p    = fq + MR;                    // 64 MB
    float2* csum = p + MR;                     // 1 MB

    // 1) QKV projection (WMMA bf16x3)
    qkv_gemm_kernel<<<dim3(M_TOT / 32, D_LEN / 128, 3), 256, 0, stream>>>(
        x, Wq, Wk, Wv, q, kk, vv);

    // 2) forward FFTs + binding
    fft_fwd_kernel<<<M_TOT, 128, 0, stream>>>(q, kk, vv, fq, p);

    // 3-5) causal cumulative sum over S (blocked scan, in place on p)
    chunk_sum_kernel<<<(B_LEN * NCHUNK * D_LEN + 255) / 256, 256, 0, stream>>>(p, csum);
    chunk_scan_kernel<<<(B_LEN * D_LEN + 255) / 256, 256, 0, stream>>>(csum);
    scan_apply_kernel<<<(B_LEN * NCHUNK * D_LEN + 255) / 256, 256, 0, stream>>>(p, csum);

    // 6) unbind with conj(fq) + inverse FFT -> real output
    ifft_kernel<<<M_TOT, 128, 0, stream>>>(p, fq, out);
}
